// SFIGF_44478681318212
// MI455X (gfx1250) — matrix operand order repaired and checked by reading, not compiled, not run on hardware
//
#include <hip/hip_runtime.h>

// ---------------------------------------------------------------------------
// SFIGF forward for gfx1250 (MI455X).
// 1x1 convs: LDS-staged (async-to-LDS when available) f16 WMMA, fp32 accum.
// Each 128-thread block: 32 Cout rows x 64 positions; each wave: M=32 x N=16
// (2 accumulators, B fragment reused -> 2 v_wmma per K-chunk of 32).
// ---------------------------------------------------------------------------

typedef __attribute__((ext_vector_type(16))) _Float16 v16h;
typedef __attribute__((ext_vector_type(8)))  float    v8f;
typedef __attribute__((ext_vector_type(4)))  int      i4vec;

__device__ __forceinline__ float gelu_erf(float x) {
    return 0.5f * x * (1.0f + erff(x * 0.7071067811865475f));
}

#define LDS_PITCH 68   // 64 cols + 4 pad: keeps 16B alignment, spreads banks

// ---------------- WMMA pointwise conv (1x1), NCHW, batch via grid.z ----------
// Y[b,co,p] = sum_ci W[co,ci]*X[b,ci,p] + bias[co]  (+ optional erf-GELU)
// Requires Cin%32==0, Cout%32==0, HW%64==0. Block = 128 threads (4 waves).
__global__ void k_conv1x1_wmma(const float* __restrict__ X,
                               const float* __restrict__ Wt,
                               const float* __restrict__ bias,
                               float* __restrict__ Y,
                               int Cin, int Cout, int HW, int do_gelu) {
    __shared__ float tile[32 * LDS_PITCH];          // 32 K-rows x 64 positions

    const int t    = threadIdx.x;                   // 0..127
    const int lane = t & 31;
    const int wid  = t >> 5;
    const int m    = lane & 15;                     // row/col-in-tile selector
    const int hi   = lane >> 4;                     // half-wave selector
    const int nb   = blockIdx.x * 64;               // block position base
    const int colB = wid * 16 + m;                  // this lane's column in tile
    const int m0   = blockIdx.y * 32;               // Cout base (2 tiles of 16)
    const int b    = blockIdx.z;
    const float* Xb = X + (size_t)b * Cin * HW;
    float*       Yb = Y + (size_t)b * Cout * HW;

    v8f acc0 = {}, acc1 = {};
    for (int k0 = 0; k0 < Cin; k0 += 32) {
        // ---- stage X[k0..k0+31][nb..nb+63] into LDS (coalesced 128b) ----
#pragma unroll
        for (int i = 0; i < 4; ++i) {
            int e  = t + i * 128;                   // vec4 index, 0..511
            int kk = e >> 4;                        // K row (16 vec4 per row)
            int c4 = (e & 15) << 2;                 // column (multiple of 4)
            const float* gp = Xb + (size_t)(k0 + kk) * HW + nb + c4;
            float*       lp = &tile[kk * LDS_PITCH + c4];
#if __has_builtin(__builtin_amdgcn_global_load_async_to_lds_b128)
            __builtin_amdgcn_global_load_async_to_lds_b128(
                (__attribute__((address_space(1))) i4vec*)gp,
                (__attribute__((address_space(3))) i4vec*)lp, 0, 0);
#else
            *(float4*)lp = *(const float4*)gp;
#endif
        }
#if __has_builtin(__builtin_amdgcn_global_load_async_to_lds_b128)
#if __has_builtin(__builtin_amdgcn_s_wait_asynccnt)
        __builtin_amdgcn_s_wait_asynccnt(0);
#else
        asm volatile("s_wait_asynccnt 0x0" ::: "memory");
#endif
#endif
        __syncthreads();

        // ---- A fragments (weights), two Cout tiles ----
        // f16 A 16x32 layout: hi=0 -> K 0..7 & 16..23 ; hi=1 -> K 8..15 & 24..31
        const float* wrow0 = Wt + (size_t)(m0 + m) * Cin + k0;
        const float* wrow1 = wrow0 + (size_t)16 * Cin;
        v16h a0, a1;
#pragma unroll
        for (int j = 0; j < 8; ++j) {
            a0[j]     = (_Float16)wrow0[hi * 8 + j];
            a0[j + 8] = (_Float16)wrow0[16 + hi * 8 + j];
            a1[j]     = (_Float16)wrow1[hi * 8 + j];
            a1[j + 8] = (_Float16)wrow1[16 + hi * 8 + j];
        }
        // ---- B fragment from LDS: half j holds K = (j/8)*16 + hi*8 + j%8, column colB ----
        v16h bv;
#pragma unroll
        for (int j = 0; j < 16; ++j)
            bv[j] = (_Float16)tile[((j >> 3) * 16 + hi * 8 + (j & 7)) * LDS_PITCH + colB];

        acc0 = __builtin_amdgcn_wmma_f32_16x16x32_f16(false, a0, false, bv,
                                                      (short)0, acc0, false, false);
        acc1 = __builtin_amdgcn_wmma_f32_16x16x32_f16(false, a1, false, bv,
                                                      (short)0, acc1, false, false);
        __syncthreads();
    }
    // D (16x16 f32): lane col nb+colB ; VGPR r -> row m0 + hi*8 + r (+16 for tile1)
#pragma unroll
    for (int r = 0; r < 8; ++r) {
        const int row0 = m0 + hi * 8 + r;
        float v0 = acc0[r] + bias[row0];
        float v1 = acc1[r] + bias[row0 + 16];
        if (do_gelu) { v0 = gelu_erf(v0); v1 = gelu_erf(v1); }
        Yb[(size_t)row0 * HW + (nb + colB)]        = v0;
        Yb[(size_t)(row0 + 16) * HW + (nb + colB)] = v1;
    }
}

// ---------------- LayerNorm over channel dim of NCHW -------------------------
__global__ void k_ln2d(const float* __restrict__ X, const float* __restrict__ g,
                       const float* __restrict__ bt, float* __restrict__ Y,
                       int C, int HW, int nPos) {
    int t = blockIdx.x * blockDim.x + threadIdx.x;
    if (t >= nPos) return;
    int b = t / HW, p = t - b * HW;
    const float* xp = X + (size_t)b * C * HW + p;
    float s = 0.f, s2 = 0.f;
    for (int c = 0; c < C; ++c) { float v = xp[(size_t)c * HW]; s += v; s2 += v * v; }
    float mu  = s / C;
    float var = s2 / C - mu * mu;
    float rs  = rsqrtf(var + 1e-6f);
    float* yp = Y + (size_t)b * C * HW + p;
    for (int c = 0; c < C; ++c) {
        float v = (xp[(size_t)c * HW] - mu) * rs;
        yp[(size_t)c * HW] = v * g[c] + bt[c];
    }
}

// ---------------- depthwise 3x3, pad 1 ---------------------------------------
__global__ void k_dw3x3(const float* __restrict__ X, const float* __restrict__ W9,
                        const float* __restrict__ bias, float* __restrict__ Y,
                        int C, int H, int Wd, int do_gelu, size_t n) {
    size_t i = (size_t)blockIdx.x * blockDim.x + threadIdx.x;
    if (i >= n) return;
    int x  = (int)(i % Wd);
    int y  = (int)((i / Wd) % H);
    size_t bc = i / ((size_t)Wd * H);
    int ch = (int)(bc % C);
    const float* xin = X + bc * (size_t)H * Wd;
    const float* w   = W9 + (size_t)ch * 9;
    float acc = bias[ch];
#pragma unroll
    for (int ky = -1; ky <= 1; ++ky) {
        int yy = y + ky;
        if (yy < 0 || yy >= H) continue;
#pragma unroll
        for (int kx = -1; kx <= 1; ++kx) {
            int xx = x + kx;
            if (xx < 0 || xx >= Wd) continue;
            acc += w[(ky + 1) * 3 + (kx + 1)] * xin[(size_t)yy * Wd + xx];
        }
    }
    if (do_gelu) acc = gelu_erf(acc);
    Y[i] = acc;
}

// ---------------- SimpleGate: split 2C -> C, multiply halves -----------------
__global__ void k_sgate(const float* __restrict__ X, float* __restrict__ Y,
                        int C, int HW, size_t n) {
    size_t i = (size_t)blockIdx.x * blockDim.x + threadIdx.x;
    if (i >= n) return;
    size_t b    = i / ((size_t)C * HW);
    size_t rest = i - b * (size_t)C * HW;
    const float* xb = X + b * (size_t)2 * C * HW;
    Y[i] = xb[rest] * xb[rest + (size_t)C * HW];
}

// ---------------- global average pool per (b,c) ------------------------------
__global__ void k_avgpool(const float* __restrict__ X, float* __restrict__ M, int HW) {
    const float* p = X + (size_t)blockIdx.x * HW;
    float s = 0.f;
    for (int i = threadIdx.x; i < HW; i += blockDim.x) s += p[i];
    __shared__ float sm[256];
    sm[threadIdx.x] = s;
    __syncthreads();
    for (int o = 128; o > 0; o >>= 1) {
        if ((int)threadIdx.x < o) sm[threadIdx.x] += sm[threadIdx.x + o];
        __syncthreads();
    }
    if (threadIdx.x == 0) M[blockIdx.x] = sm[0] / (float)HW;
}

// ---------------- tiny SCA matvec: S[b,co] = Wsca.M[b,:] + bsca --------------
__global__ void k_sca(const float* __restrict__ M, const float* __restrict__ Wm,
                      const float* __restrict__ bias, float* __restrict__ S,
                      int C, int BC) {
    int t = blockIdx.x * blockDim.x + threadIdx.x;
    if (t >= BC) return;
    int b = t / C, co = t - b * C;
    const float* m = M + (size_t)b * C;
    const float* w = Wm + (size_t)co * C;
    float s = bias[co];
    for (int ci = 0; ci < C; ++ci) s += w[ci] * m[ci];
    S[t] = s;
}

// ---------------- in-place per-(b,c) scale -----------------------------------
__global__ void k_scale_bc(float* __restrict__ Y, const float* __restrict__ S,
                           int HW, size_t n) {
    size_t i = (size_t)blockIdx.x * blockDim.x + threadIdx.x;
    if (i >= n) return;
    Y[i] *= S[i / HW];
}

// ---------------- out = base + y*scale[c] (+extra) ---------------------------
__global__ void k_res3(const float* __restrict__ base, const float* __restrict__ y,
                       const float* __restrict__ scale, const float* __restrict__ extra,
                       float* __restrict__ out, int C, int HW, size_t n) {
    size_t i = (size_t)blockIdx.x * blockDim.x + threadIdx.x;
    if (i >= n) return;
    int ch = (int)((i / HW) % C);
    float v = base[i] + y[i] * scale[ch];
    if (extra) v += extra[i];
    out[i] = v;
}

// ---------------- 2x2 maxpool, stride 2 --------------------------------------
__global__ void k_maxpool2(const float* __restrict__ X, float* __restrict__ Y,
                           int H, int Wd, size_t n) {
    size_t i = (size_t)blockIdx.x * blockDim.x + threadIdx.x;
    if (i >= n) return;
    int Wo = Wd >> 1, Ho = H >> 1;
    int xo = (int)(i % Wo);
    int yo = (int)((i / Wo) % Ho);
    size_t bc = i / ((size_t)Wo * Ho);
    const float* p = X + bc * (size_t)H * Wd + (size_t)(2 * yo) * Wd + 2 * xo;
    float a = fmaxf(p[0], p[1]);
    float b = fmaxf(p[Wd], p[Wd + 1]);
    Y[i] = fmaxf(a, b);
}

// ---------------- channel concat ---------------------------------------------
__global__ void k_concat(const float* __restrict__ A, const float* __restrict__ B,
                         float* __restrict__ Y, int C1, int C2, int HW, size_t n) {
    size_t i = (size_t)blockIdx.x * blockDim.x + threadIdx.x;
    if (i >= n) return;
    int Ct = C1 + C2;
    int p  = (int)(i % HW);
    int c  = (int)((i / HW) % Ct);
    size_t b = i / ((size_t)HW * Ct);
    Y[i] = (c < C1) ? A[(b * C1 + c) * (size_t)HW + p]
                    : B[(b * C2 + (c - C1)) * (size_t)HW + p];
}

// ===========================================================================
// Host side
// ===========================================================================
namespace {

struct DownP { const float *b, *w; };
struct DigP  { const float *b1,*b2,*b3,*w1,*w2,*w3; };
struct NafP  { const float *b1,*b2,*b3,*b4,*b5,*beta,*bsca,*gamma,
                           *n1b,*n1w,*n2b,*n2w,*w1,*w2,*w3,*w4,*w5,*wsca; };
struct SeqP  { NafP naf; const float *pb, *pw; };

struct Ctx {
    hipStream_t s;
    float *A, *D, *Bw, *Cw, *mb, *sb;   // arena buffers
};

inline dim3 eg(size_t n) { return dim3((unsigned)((n + 255) / 256)); }

void run_conv(const Ctx& cx, const float* X, const float* W, const float* bias,
              float* Y, int Cin, int Cout, int HW, int gelu) {
    dim3 g(HW / 64, Cout / 32, 2);
    k_conv1x1_wmma<<<g, 128, 0, cx.s>>>(X, W, bias, Y, Cin, Cout, HW, gelu);
}

// _seq: concat -> 1x1+GELU -> NAFBlock; out = NAF(...) + resid
void run_seq(const Ctx& cx, const SeqP& P,
             const float* xa, int Ca, const float* xb, int Cb,
             const float* resid, float* out, int c, int H, int W) {
    const int Bn = 2, HW = H * W, Cin = Ca + Cb;
    size_t nCat = (size_t)Bn * Cin * HW;
    size_t nPos = (size_t)Bn * HW;
    size_t nC   = (size_t)Bn * c * HW;
    size_t n2C  = (size_t)Bn * 2 * c * HW;

    k_concat<<<eg(nCat), 256, 0, cx.s>>>(xa, xb, cx.A, Ca, Cb, HW, nCat);
    run_conv(cx, cx.A, P.pw, P.pb, cx.Bw, Cin, c, HW, 1);              // h = gelu(pw(cat))
    // --- NAF block ---
    k_ln2d<<<eg(nPos), 256, 0, cx.s>>>(cx.Bw, P.naf.n1w, P.naf.n1b, cx.Cw, c, HW, (int)nPos);
    run_conv(cx, cx.Cw, P.naf.w1, P.naf.b1, cx.A, c, 2 * c, HW, 0);    // c -> 2c
    k_dw3x3<<<eg(n2C), 256, 0, cx.s>>>(cx.A, P.naf.w2, P.naf.b2, cx.D, 2 * c, H, W, 0, n2C);
    k_sgate<<<eg(nC), 256, 0, cx.s>>>(cx.D, cx.Cw, c, HW, nC);         // SimpleGate
    k_avgpool<<<dim3(Bn * c), 256, 0, cx.s>>>(cx.Cw, cx.mb, HW);       // SCA pool
    k_sca<<<eg((size_t)Bn * c), 256, 0, cx.s>>>(cx.mb, P.naf.wsca, P.naf.bsca, cx.sb, c, Bn * c);
    k_scale_bc<<<eg(nC), 256, 0, cx.s>>>(cx.Cw, cx.sb, HW, nC);        // y *= sca
    run_conv(cx, cx.Cw, P.naf.w3, P.naf.b3, cx.A, c, c, HW, 0);
    k_res3<<<eg(nC), 256, 0, cx.s>>>(cx.Bw, cx.A, P.naf.beta, nullptr, cx.D, c, HW, nC); // yy
    k_ln2d<<<eg(nPos), 256, 0, cx.s>>>(cx.D, P.naf.n2w, P.naf.n2b, cx.Bw, c, HW, (int)nPos);
    run_conv(cx, cx.Bw, P.naf.w4, P.naf.b4, cx.A, c, 2 * c, HW, 0);    // FFN expand
    k_sgate<<<eg(nC), 256, 0, cx.s>>>(cx.A, cx.Bw, c, HW, nC);
    run_conv(cx, cx.Bw, P.naf.w5, P.naf.b5, cx.Cw, c, c, HW, 0);
    k_res3<<<eg(nC), 256, 0, cx.s>>>(cx.D, cx.Cw, P.naf.gamma, resid, out, c, HW, nC);
}

// _down: maxpool2(gelu(1x1 c->2c))
void run_down(const Ctx& cx, const DownP& P, const float* x, int c, int H, int W, float* out) {
    int HW = H * W;
    run_conv(cx, x, P.w, P.b, cx.A, c, 2 * c, HW, 1);
    size_t nOut = (size_t)2 * 2 * c * (H / 2) * (W / 2);
    k_maxpool2<<<eg(nOut), 256, 0, cx.s>>>(cx.A, out, H, W, nOut);
}

// _dig: cat -> 1x1(2c->c)+G -> dw3x3+G -> 1x1(c->2c)+G -> maxpool2
void run_dig(const Ctx& cx, const DigP& P, const float* xa, const float* xb,
             int c, int H, int W, float* out) {
    int HW = H * W;
    size_t nCat = (size_t)2 * 2 * c * HW;
    size_t nC   = (size_t)2 * c * HW;
    k_concat<<<eg(nCat), 256, 0, cx.s>>>(xa, xb, cx.A, c, c, HW, nCat);
    run_conv(cx, cx.A, P.w1, P.b1, cx.Bw, 2 * c, c, HW, 1);
    k_dw3x3<<<eg(nC), 256, 0, cx.s>>>(cx.Bw, P.w2, P.b2, cx.Cw, c, H, W, 1, nC);
    run_conv(cx, cx.Cw, P.w3, P.b3, cx.A, c, 2 * c, HW, 1);
    size_t nOut = (size_t)2 * 2 * c * (H / 2) * (W / 2);
    k_maxpool2<<<eg(nOut), 256, 0, cx.s>>>(cx.A, out, H, W, nOut);
}

} // namespace

extern "C" void kernel_launch(void* const* d_in, const int* in_sizes, int n_in,
                              void* d_out, int out_size, void* d_ws, size_t ws_size,
                              hipStream_t stream) {
    (void)in_sizes; (void)n_in; (void)out_size; (void)ws_size;

    const float* inp = (const float*)d_in[0];
    const float* gde = (const float*)d_in[1];
    const float* ipg = (const float*)d_in[2];

    // ---- param leaves: pytree order (keys sorted alphabetically, depth-first)
    int idx = 3;
    auto nx = [&]() { return (const float*)d_in[idx++]; };
    auto rdDown = [&](DownP& p) { p.b = nx(); p.w = nx(); };
    auto rdDig  = [&](DigP& p)  { p.b1=nx(); p.b2=nx(); p.b3=nx(); p.w1=nx(); p.w2=nx(); p.w3=nx(); };
    auto rdNaf  = [&](NafP& p)  { p.b1=nx(); p.b2=nx(); p.b3=nx(); p.b4=nx(); p.b5=nx();
                                  p.beta=nx(); p.bsca=nx(); p.gamma=nx();
                                  p.n1b=nx(); p.n1w=nx(); p.n2b=nx(); p.n2w=nx();
                                  p.w1=nx(); p.w2=nx(); p.w3=nx(); p.w4=nx(); p.w5=nx(); p.wsca=nx(); };
    auto rdSeq  = [&](SeqP& p)  { rdNaf(p.naf); p.pb = nx(); p.pw = nx(); };

    DownP dg1, dg2, dg3, di1, di2, di3;
    DigP  dig1, dig2, dig3;
    SeqP  ng, ng2, ng3, ng4, ni, ni2, ni3, ni4;
    rdDown(dg1); rdDown(dg2); rdDown(dg3);
    rdDown(di1); rdDown(di2); rdDown(di3);
    rdDig(dig1); rdDig(dig2); rdDig(dig3);
    rdSeq(ng);  rdSeq(ng2); rdSeq(ng3); rdSeq(ng4);
    rdSeq(ni);  rdSeq(ni2); rdSeq(ni3); rdSeq(ni4);

    // ---- workspace arena (~157 MB): 2x "2c" buffers + 2x "c" buffers + SCA scratch
    float* ws = (float*)d_ws;
    Ctx cx;
    cx.s  = stream;
    cx.A  = ws;                     // 13,107,200 floats (2*64*320*320)
    cx.D  = cx.A  + 13107200;       // 13,107,200
    cx.Bw = cx.D  + 13107200;       //  6,553,600
    cx.Cw = cx.Bw + 6553600;        //  6,553,600
    cx.mb = cx.Cw + 6553600;        //    512 (avgpool means)
    cx.sb = cx.mb + 512;            //    512 (sca scales)

    // ---- output offsets (floats, return order) ----
    float* out = (float*)d_out;
    const size_t o_inp1 = 0,          o_guide1 = 6553600;
    const size_t o_inp2 = 13107200,   o_guide2 = 16384000;
    const size_t o_inp3 = 19660800,   o_guide3 = 21299200;
    const size_t o_inp4 = 22937600,   o_guide4 = 23756800;
    const size_t o_ipg  = 24576000;
    const size_t o_ig2  = 31129600,   o_ig3 = 34406400, o_ig4 = 36044800;

    // inp_guide passthrough
    (void)hipMemcpyAsync(out + o_ipg, ipg, (size_t)6553600 * sizeof(float),
                         hipMemcpyDeviceToDevice, stream);

    // ---- level 1 (c=32, 320x320) ----
    run_seq(cx, ni, inp, 32, ipg, 32, inp, out + o_inp1, 32, 320, 320);
    run_seq(cx, ng, gde, 32, ipg, 32, gde, out + o_guide1, 32, 320, 320);
    run_down(cx, di1, out + o_inp1,   32, 320, 320, out + o_inp2);
    run_down(cx, dg1, out + o_guide1, 32, 320, 320, out + o_guide2);
    run_dig(cx, dig1, out + o_inp1, out + o_guide1, 32, 320, 320, out + o_ig2);

    // ---- level 2 (c=64, 160x160) ----
    run_seq(cx, ni2, out + o_inp2,   64, out + o_ig2, 64, out + o_inp2,   out + o_inp2,   64, 160, 160);
    run_seq(cx, ng2, out + o_guide2, 64, out + o_ig2, 64, out + o_guide2, out + o_guide2, 64, 160, 160);
    run_down(cx, di2, out + o_inp2,   64, 160, 160, out + o_inp3);
    run_down(cx, dg2, out + o_guide2, 64, 160, 160, out + o_guide3);
    run_dig(cx, dig2, out + o_inp2, out + o_guide2, 64, 160, 160, out + o_ig3);

    // ---- level 3 (c=128, 80x80) ----
    run_seq(cx, ni3, out + o_inp3,   128, out + o_ig3, 128, out + o_inp3,   out + o_inp3,   128, 80, 80);
    run_seq(cx, ng3, out + o_guide3, 128, out + o_ig3, 128, out + o_guide3, out + o_guide3, 128, 80, 80);
    run_down(cx, di3, out + o_inp3,   128, 80, 80, out + o_inp4);
    run_down(cx, dg3, out + o_guide3, 128, 80, 80, out + o_guide4);
    run_dig(cx, dig3, out + o_inp3, out + o_guide3, 128, 80, 80, out + o_ig4);

    // ---- level 4 (c=256, 40x40) ----
    run_seq(cx, ni4, out + o_inp4,   256, out + o_ig4, 256, out + o_inp4,   out + o_inp4,   256, 40, 40);
    run_seq(cx, ng4, out + o_guide4, 256, out + o_ig4, 256, out + o_guide4, out + o_guide4, 256, 40, 40);
}
